// PatchGNP_50852412785143
// MI455X (gfx1250) — matrix operand order repaired and checked by reading, not compiled, not run on hardware
//
#include <hip/hip_runtime.h>
#include <hip/hip_bf16.h>

// ---------------------------------------------------------------------------
// PatchGNP on MI455X (gfx1250): bf16 WMMA everywhere K>=16, f32 accumulate.
// Sorted `batch` => contiguous per-patch row ranges => segment ops are dense
// per-patch GEMMs. Weights pre-shuffled into WMMA B-fragment order (one 32B
// load per fragment). Tails use unconditional loads into slack memory with
// value-masking (v_cndmask), never conditional loads.
// ---------------------------------------------------------------------------

#define NPTS 65536
#define PP   128
#define DXD  3
#define WID  128
#define CC   16
#define HH   128
#define OUTD 64
#define LBLK 2

// slack (elements) appended to streamed buffers so tail fragments may over-read
#define SLACK 8192

typedef __attribute__((ext_vector_type(16))) __bf16 v16bf;
typedef __attribute__((ext_vector_type(8)))  __bf16 v8bf;
typedef __attribute__((ext_vector_type(8)))  float  v8f;

__device__ __forceinline__ float gelu_f(float x) {
    float x3 = x * x * x;
    return 0.5f * x * (1.0f + tanhf(0.7978845608028654f * (x + 0.044715f * x3)));
}

__device__ __forceinline__ v8f wmma_bf16(v16bf a, v16bf b, v8f c) {
    return __builtin_amdgcn_wmma_f32_16x16x32_bf16(
        false, a, false, b, (short)0, c, false, false);
}

// ---------------- fast (unguarded) fragment loaders ------------------------
// A 16x32: lane<16 -> M=lane, K {0..15}; lane>=16 -> K
// {16..31}. Row-major A => two contiguous 16B runs per lane.
__device__ __forceinline__ v16bf load_a_row32(const __bf16* base, int rstride,
                                              int lane) {
    int half = lane >> 4, m = lane & 15;
    const __bf16* q = base + (size_t)m * rstride + half * 16;
    v8bf lo = *(const v8bf*)q;
    v8bf hi = *(const v8bf*)(q + 8);
    v16bf a;
#pragma unroll
    for (int i = 0; i < 8; ++i) { a[i] = lo[i]; a[8 + i] = hi[i]; }
    return a;
}

// A with K=16 (upper K half zero): two 16B loads.
__device__ __forceinline__ v16bf load_a_row16(const __bf16* base, int rstride,
                                              int lane) {
    int half = lane >> 4, m = lane & 15;
    v8bf lo = *(const v8bf*)(base + (size_t)m * rstride);
    v8bf hi = *(const v8bf*)(base + (size_t)m * rstride + 8);
    v16bf a;
#pragma unroll
    for (int i = 0; i < 8; ++i) {
        a[i] = half ? (__bf16)0.0f : lo[i];
        a[8 + i] = half ? (__bf16)0.0f : hi[i];
    }
    return a;
}

// B 32x16 from row-major memory (dynamic operands only): 16 u16 loads/lane.
__device__ __forceinline__ v16bf load_b_row(const __bf16* base, int ldb,
                                            int lane) {
    int half = lane >> 4, n = lane & 15;
    v16bf b;
#pragma unroll
    for (int r = 0; r < 8; ++r) {
        int k = half * 16 + 2 * r;
        b[2 * r]     = base[(size_t)k * ldb + n];
        b[2 * r + 1] = base[(size_t)(k + 1) * ldb + n];
    }
    return b;
}

// B with K=16: clamp K into the valid 16 rows (no OOB), mask upper half to 0.
// Unconditional loads + v_cndmask; no EXEC divergence.
__device__ __forceinline__ v16bf load_b_row_k16(const __bf16* base, int ldb,
                                                int lane) {
    int half = lane >> 4, n = lane & 15;
    v16bf b;
#pragma unroll
    for (int r = 0; r < 8; ++r) {
        int kc = 2 * r;                       // (half*16 + 2r) & 15
        __bf16 v0 = base[(size_t)kc * ldb + n];
        __bf16 v1 = base[(size_t)(kc + 1) * ldb + n];
        b[2 * r]     = half ? (__bf16)0.0f : v0;
        b[2 * r + 1] = half ? (__bf16)0.0f : v1;
    }
    return b;
}

// B from pre-packed weight: [tile][lane][16] => one 32B per-lane load.
__device__ __forceinline__ v16bf load_b_packed(const __bf16* tile_base,
                                               int lane) {
    return *(const v16bf*)(tile_base + (size_t)lane * 16);
}

// ---------------- masked loaders (head/tail chunks only) -------------------
// Loads are UNCONDITIONAL (buffers carry SLACK elements past the end); only
// the loaded VALUE is masked => clause-batched loads + cndmask, no branches.
__device__ __forceinline__ v16bf load_a_mask(const __bf16* base, int rstride,
                                             int kstride, int mlim, int klim,
                                             int lane) {
    int half = lane >> 4, m = lane & 15;
    bool mok = m < mlim;
    v16bf a;
#pragma unroll
    for (int i = 0; i < 8; ++i) {
        int k = half * 16 + 2 * i;
        __bf16 e0 = base[(size_t)m * rstride + (size_t)k * kstride];
        __bf16 e1 = base[(size_t)m * rstride + (size_t)(k + 1) * kstride];
        a[2 * i]     = (mok && (k < klim))     ? e0 : (__bf16)0.0f;
        a[2 * i + 1] = (mok && (k + 1 < klim)) ? e1 : (__bf16)0.0f;
    }
    return a;
}

__device__ __forceinline__ v16bf load_b_mask(const __bf16* base, int kstride,
                                             int nstride, int klim, int lane) {
    int half = lane >> 4, n = lane & 15;
    v16bf b;
#pragma unroll
    for (int r = 0; r < 8; ++r) {
        int k = half * 16 + 2 * r;
        __bf16 e0 = base[(size_t)k * kstride + (size_t)n * nstride];
        __bf16 e1 = base[(size_t)(k + 1) * kstride + (size_t)n * nstride];
        b[2 * r]     = (k < klim)     ? e0 : (__bf16)0.0f;
        b[2 * r + 1] = (k + 1 < klim) ? e1 : (__bf16)0.0f;
    }
    return b;
}

// ---------------- C/D tile stores ------------------------------------------
__device__ __forceinline__ void store_tile_bf16_full(__bf16* base, int rstride,
                                                     int lane, v8f d) {
    int half = lane >> 4, n = lane & 15;
#pragma unroll
    for (int r = 0; r < 8; ++r)
        base[(size_t)(half * 8 + r) * rstride + n] = (__bf16)d[r];
}

__device__ __forceinline__ void store_tile_bf16(__bf16* base, int rstride,
                                                int mlim, int lane, v8f d) {
    int half = lane >> 4, n = lane & 15;
#pragma unroll
    for (int r = 0; r < 8; ++r) {
        int m = half * 8 + r;
        if (m < mlim) base[(size_t)m * rstride + n] = (__bf16)d[r];
    }
}

__device__ __forceinline__ void store_tile_f32(float* base, int rstride,
                                               int lane, v8f d) {
    int half = lane >> 4, n = lane & 15;
#pragma unroll
    for (int r = 0; r < 8; ++r)
        base[(size_t)(half * 8 + r) * rstride + n] = d[r];
}

// ---------------------------------------------------------------------------
// Kernel 1: patch row ranges from sorted batch (binary search) + inv counts.
// ---------------------------------------------------------------------------
__global__ void k_ranges(const int* __restrict__ batch, int* __restrict__ start,
                         float* __restrict__ inv) {
    int t = threadIdx.x;
    if (t <= PP) {
        int lo = 0, hi = NPTS;
        while (lo < hi) {
            int mid = (lo + hi) >> 1;
            if (batch[mid] < t) lo = mid + 1; else hi = mid;
        }
        start[t] = lo;
    }
    __syncthreads();
    if (t < PP) {
        float c = (float)(start[t + 1] - start[t]);
        inv[t]  = 1.0f / fmaxf(c, 1.0f);
    }
}

// ---------------------------------------------------------------------------
// Kernel 2: pack f32 row-major weight (K x Nn) into bf16 B-fragment order:
// dst[(kt*ntiles + nt)*32 + lane][i], lane = (kk>=16)*16 + nn, i = kk & 15.
// Batched over blockIdx.y matrices.
// ---------------------------------------------------------------------------
__global__ void k_pack_b(const float* __restrict__ src, __bf16* __restrict__ dst,
                         int K, int Nn) {
    size_t mat_off = (size_t)blockIdx.y * K * Nn;
    int idx = blockIdx.x * blockDim.x + threadIdx.x;
    if (idx >= K * Nn) return;
    int k = idx / Nn, n = idx - k * Nn;
    int kt = k >> 5, kk = k & 31;
    int nt = n >> 4, nn = n & 15;
    int lane = ((kk >> 4) << 4) + nn;
    int i    = kk & 15;
    int ntiles = Nn >> 4;
    dst[mat_off + ((((size_t)kt * ntiles) + nt) * 32 + lane) * 16 + i] =
        (__bf16)src[mat_off + idx];
}

// ---------------------------------------------------------------------------
// Kernel 3: lift v0 = x @ lift_w + lift_b   (K=3: VALU)
// ---------------------------------------------------------------------------
__global__ void k_lift(const float* __restrict__ x, const float* __restrict__ lw,
                       const float* __restrict__ lb, __bf16* __restrict__ v) {
    int i = blockIdx.x * blockDim.x + threadIdx.x;
    if (i >= NPTS * WID) return;
    int n = i >> 7, w = i & (WID - 1);
    float a = x[n * 3 + 0] * lw[0 * WID + w] +
              x[n * 3 + 1] * lw[1 * WID + w] +
              x[n * 3 + 2] * lw[2 * WID + w] + lb[w];
    v[i] = (__bf16)a;
}

// ---------------------------------------------------------------------------
// Kernel 4: kernel MLP, one thread per point, weights in LDS.
// Output strides (sn, sc) let phi be stored (N,C) and psi transposed (C,N).
// ---------------------------------------------------------------------------
__global__ void k_mlp(const float* __restrict__ x, const float* __restrict__ w1,
                      const float* __restrict__ b1, const float* __restrict__ w2,
                      const float* __restrict__ b2, __bf16* __restrict__ out,
                      int sn, int sc) {
    __shared__ float sw1[DXD * HH];
    __shared__ float sb1[HH];
    __shared__ float sw2[HH * CC];
    __shared__ float sb2[CC];
    for (int i = threadIdx.x; i < DXD * HH; i += blockDim.x) sw1[i] = w1[i];
    for (int i = threadIdx.x; i < HH;       i += blockDim.x) sb1[i] = b1[i];
    for (int i = threadIdx.x; i < HH * CC;  i += blockDim.x) sw2[i] = w2[i];
    for (int i = threadIdx.x; i < CC;       i += blockDim.x) sb2[i] = b2[i];
    __syncthreads();

    int n = blockIdx.x * blockDim.x + threadIdx.x;
    if (n >= NPTS) return;
    float x0 = x[n * 3 + 0], x1 = x[n * 3 + 1], x2 = x[n * 3 + 2];
    float acc[CC];
#pragma unroll
    for (int c = 0; c < CC; ++c) acc[c] = sb2[c];
    for (int h = 0; h < HH; ++h) {
        float hv = gelu_f(x0 * sw1[0 * HH + h] + x1 * sw1[1 * HH + h] +
                          x2 * sw1[2 * HH + h] + sb1[h]);
#pragma unroll
        for (int c = 0; c < CC; ++c) acc[c] += hv * sw2[h * CC + c];
    }
#pragma unroll
    for (int c = 0; c < CC; ++c)
        out[(size_t)n * sn + (size_t)c * sc] = (__bf16)acc[c];
}

// ---------------------------------------------------------------------------
// Kernel 5: s[p] = inv[p] * Psi_p^T @ V_p.  psiT stored (C, NPTS) so the A
// operand is a row-major fast load. Head chunk re-aligns j to 32; head/tail
// use masked (branch-free) loaders.
// ---------------------------------------------------------------------------
__global__ void k_seg_outer(const __bf16* __restrict__ psiT,
                            const __bf16* __restrict__ v,
                            const int* __restrict__ start,
                            const float* __restrict__ inv,
                            __bf16* __restrict__ s) {
    int p    = blockIdx.x;
    int wave = threadIdx.x >> 5;
    int lane = threadIdx.x & 31;
    int n0   = wave * 16;
    int j0   = start[p], j1 = start[p + 1];
    v8f acc = {};

    int ja = (j0 + 31) & ~31;                 // aligned start for fast path
    int head_end = (ja < j1) ? ja : j1;
    if (j0 < head_end) {
        int klim = head_end - j0;
        v16bf a = load_a_mask(psiT + j0, NPTS, 1, 16, klim, lane);
        v16bf b = load_b_mask(v + (size_t)j0 * WID + n0, WID, 1, klim, lane);
        acc = wmma_bf16(a, b, acc);
    }
    int j = ja;
    for (; j + 32 <= j1; j += 32) {
        __builtin_prefetch(v + (size_t)(j + 32) * WID + n0, 0, 1);
        v16bf a = load_a_row32(psiT + j, NPTS, lane);
        v16bf b = load_b_row(v + (size_t)j * WID + n0, WID, lane);
        acc = wmma_bf16(a, b, acc);
    }
    if (j >= ja && j < j1) {
        int klim = j1 - j;
        v16bf a = load_a_mask(psiT + j, NPTS, 1, 16, klim, lane);
        v16bf b = load_b_mask(v + (size_t)j * WID + n0, WID, 1, klim, lane);
        acc = wmma_bf16(a, b, acc);
    }
    float sc = inv[p];
#pragma unroll
    for (int r = 0; r < 8; ++r) acc[r] *= sc;
    store_tile_bf16_full(s + (size_t)p * CC * WID + n0, WID, lane, acc);
}

// ---------------------------------------------------------------------------
// Kernel 6: t[:,c,:] = s[:,c,:] @ Wc[c]  (128x128x128 per channel).
// A fast row loads; B from packed weights (one 32B load per fragment).
// ---------------------------------------------------------------------------
__global__ void k_chan_gemm(const __bf16* __restrict__ s,
                            const __bf16* __restrict__ wcP,
                            __bf16* __restrict__ t) {
    int pm0  = blockIdx.x * 16;
    int c    = blockIdx.y;
    int wave = threadIdx.x >> 5;
    int lane = threadIdx.x & 31;
    const __bf16* A  = s + (size_t)pm0 * (CC * WID) + c * WID;
    const __bf16* Bp = wcP + (size_t)c * WID * WID;   // packed, ntiles = 8
    v8f acc = {};
#pragma unroll
    for (int k0 = 0; k0 < WID; k0 += 32) {
        v16bf a = load_a_row32(A + k0, CC * WID, lane);
        v16bf b = load_b_packed(Bp + (((size_t)(k0 >> 5) * 8 + wave) * 32) * 16,
                                lane);
        acc = wmma_bf16(a, b, acc);
    }
    store_tile_bf16_full(t + (size_t)pm0 * (CC * WID) + c * WID + wave * 16,
                         CC * WID, lane, acc);
}

// ---------------------------------------------------------------------------
// Kernel 7: v' = Phi_p @ t[p] (K=16) + v @ loc_w (K=128) + loc_b [, gelu].
// t[p] B-fragment hoisted out of the row-tile loop; packed loc_w B; the one
// partial row tile per patch uses masked loads + guarded store.
// ---------------------------------------------------------------------------
__global__ void k_block_update(const __bf16* __restrict__ phi,
                               const __bf16* __restrict__ t,
                               const __bf16* __restrict__ vcur,
                               const __bf16* __restrict__ locwP,
                               const float* __restrict__ locb,
                               const int* __restrict__ start,
                               __bf16* __restrict__ vnext, int apply_gelu) {
    int p    = blockIdx.x;
    int nt   = blockIdx.y;
    int n0   = nt * 16;
    int wave = threadIdx.x >> 5;
    int lane = threadIdx.x & 31;
    int j0   = start[p];
    int np   = start[p + 1] - j0;
    float bias = locb[n0 + (lane & 15)];

    // t[p] slice: B fragment with K=16, reused by every row tile.
    v16bf bt = load_b_row_k16(t + (size_t)p * (CC * WID) + n0, WID, lane);

    int nfull = np >> 4;
    for (int rt = wave; rt < nfull; rt += 8) {
        int row0 = j0 + rt * 16;
        __builtin_prefetch(vcur + (size_t)(row0 + 128) * WID, 0, 1);
        v8f acc = {};
        v16bf ap = load_a_row16(phi + (size_t)row0 * CC, CC, lane);
        acc = wmma_bf16(ap, bt, acc);
#pragma unroll
        for (int k0 = 0; k0 < WID; k0 += 32) {
            v16bf av = load_a_row32(vcur + (size_t)row0 * WID + k0, WID, lane);
            v16bf bw = load_b_packed(
                locwP + (((size_t)(k0 >> 5) * 8 + nt) * 32) * 16, lane);
            acc = wmma_bf16(av, bw, acc);
        }
#pragma unroll
        for (int r = 0; r < 8; ++r) {
            float val = acc[r] + bias;
            acc[r] = apply_gelu ? gelu_f(val) : val;
        }
        store_tile_bf16_full(vnext + (size_t)row0 * WID + n0, WID, lane, acc);
    }
    int rem = np & 15;
    if (rem && (nfull % 8) == wave) {          // masked tail tile
        int row0 = j0 + nfull * 16;
        v8f acc = {};
        v16bf ap = load_a_mask(phi + (size_t)row0 * CC, CC, 1, rem, CC, lane);
        acc = wmma_bf16(ap, bt, acc);
#pragma unroll
        for (int k0 = 0; k0 < WID; k0 += 32) {
            v16bf av = load_a_mask(vcur + (size_t)row0 * WID + k0, WID, 1, rem,
                                   32, lane);
            v16bf bw = load_b_packed(
                locwP + (((size_t)(k0 >> 5) * 8 + nt) * 32) * 16, lane);
            acc = wmma_bf16(av, bw, acc);
        }
#pragma unroll
        for (int r = 0; r < 8; ++r) {
            float val = acc[r] + bias;
            acc[r] = apply_gelu ? gelu_f(val) : val;
        }
        store_tile_bf16(vnext + (size_t)row0 * WID + n0, WID, rem, lane, acc);
    }
}

// ---------------------------------------------------------------------------
// Kernel 8: per-patch mean pooling (thread = column, coalesced).
// ---------------------------------------------------------------------------
__global__ void k_pool(const __bf16* __restrict__ v, const int* __restrict__ start,
                       const float* __restrict__ inv,
                       __bf16* __restrict__ pooled) {
    int p = blockIdx.x, w = threadIdx.x;
    int j0 = start[p], j1 = start[p + 1];
    float s = 0.0f;
    for (int j = j0; j < j1; ++j) s += (float)v[(size_t)j * WID + w];
    pooled[(size_t)p * WID + w] = (__bf16)(s * inv[p]);
}

// ---------------------------------------------------------------------------
// Kernel 9: h = gelu(pooled @ p1_w + p1_b)  (128x128 @ 128x256), 1 wave/tile.
// ---------------------------------------------------------------------------
__global__ void k_proj1(const __bf16* __restrict__ pooled,
                        const __bf16* __restrict__ p1P,
                        const float* __restrict__ b, __bf16* __restrict__ h) {
    int m0 = blockIdx.x * 16, nt = blockIdx.y;
    int lane = threadIdx.x & 31;
    v8f acc = {};
#pragma unroll
    for (int k0 = 0; k0 < WID; k0 += 32) {
        v16bf a = load_a_row32(pooled + (size_t)m0 * WID + k0, WID, lane);
        v16bf bb = load_b_packed(
            p1P + (((size_t)(k0 >> 5) * 16 + nt) * 32) * 16, lane);  // ntiles=16
        acc = wmma_bf16(a, bb, acc);
    }
    float bias = b[nt * 16 + (lane & 15)];
#pragma unroll
    for (int r = 0; r < 8; ++r) acc[r] = gelu_f(acc[r] + bias);
    store_tile_bf16_full(h + (size_t)m0 * (2 * WID) + nt * 16, 2 * WID, lane,
                         acc);
}

// ---------------------------------------------------------------------------
// Kernel 10: out = h @ p2_w + p2_b  (128x256 @ 256x64), f32 output.
// ---------------------------------------------------------------------------
__global__ void k_proj2(const __bf16* __restrict__ h,
                        const __bf16* __restrict__ p2P,
                        const float* __restrict__ b, float* __restrict__ out) {
    int m0 = blockIdx.x * 16, nt = blockIdx.y;
    int lane = threadIdx.x & 31;
    v8f acc = {};
#pragma unroll
    for (int k0 = 0; k0 < 2 * WID; k0 += 32) {
        v16bf a = load_a_row32(h + (size_t)m0 * (2 * WID) + k0, 2 * WID, lane);
        v16bf bb = load_b_packed(
            p2P + (((size_t)(k0 >> 5) * 4 + nt) * 32) * 16, lane);   // ntiles=4
        acc = wmma_bf16(a, bb, acc);
    }
    float bias = b[nt * 16 + (lane & 15)];
#pragma unroll
    for (int r = 0; r < 8; ++r) acc[r] += bias;
    store_tile_f32(out + (size_t)m0 * OUTD + nt * 16, OUTD, lane, acc);
}

// ---------------------------------------------------------------------------
// Host launcher
// ---------------------------------------------------------------------------
static inline size_t align_up(size_t x) { return (x + 255) & ~(size_t)255; }

extern "C" void kernel_launch(void* const* d_in, const int* in_sizes, int n_in,
                              void* d_out, int out_size, void* d_ws, size_t ws_size,
                              hipStream_t stream) {
    (void)in_sizes; (void)n_in; (void)out_size; (void)ws_size;

    const float* x      = (const float*)d_in[0];
    const int*   batch  = (const int*)d_in[1];
    const float* lift_w = (const float*)d_in[3];
    const float* lift_b = (const float*)d_in[4];
    const float* phi_w1 = (const float*)d_in[5];
    const float* phi_b1 = (const float*)d_in[6];
    const float* phi_w2 = (const float*)d_in[7];
    const float* phi_b2 = (const float*)d_in[8];
    const float* psi_w1 = (const float*)d_in[9];
    const float* psi_b1 = (const float*)d_in[10];
    const float* psi_w2 = (const float*)d_in[11];
    const float* psi_b2 = (const float*)d_in[12];
    const float* Wc     = (const float*)d_in[13];
    const float* loc_w  = (const float*)d_in[14];
    const float* loc_b  = (const float*)d_in[15];
    const float* p1_w   = (const float*)d_in[16];
    const float* p1_b   = (const float*)d_in[17];
    const float* p2_w   = (const float*)d_in[18];
    const float* p2_b   = (const float*)d_in[19];
    float* out = (float*)d_out;

    // --- workspace carve-out (~43 MB, SLACK pads for masked tail over-reads)
    char* wsp = (char*)d_ws;
    size_t off = 0;
    auto carve = [&](size_t bytes) -> void* {
        void* pr = (void*)(wsp + off);
        off = align_up(off + bytes);
        return pr;
    };
    int*    start  = (int*)carve((PP + 1) * sizeof(int));
    float*  inv    = (float*)carve(PP * sizeof(float));
    __bf16* v_a    = (__bf16*)carve(((size_t)NPTS * WID + SLACK) * 2);
    __bf16* v_b    = (__bf16*)carve(((size_t)NPTS * WID + SLACK) * 2);
    __bf16* phiB   = (__bf16*)carve(((size_t)LBLK * NPTS * CC + SLACK) * 2); // (N,C)
    __bf16* psiT   = (__bf16*)carve(((size_t)LBLK * CC * NPTS + SLACK) * 2); // (C,N)
    __bf16* sbuf   = (__bf16*)carve((size_t)PP * CC * WID * 2);
    __bf16* tbuf   = (__bf16*)carve((size_t)PP * CC * WID * 2);
    __bf16* pooled = (__bf16*)carve((size_t)PP * WID * 2);
    __bf16* hbuf   = (__bf16*)carve((size_t)PP * 2 * WID * 2);
    __bf16* WcP    = (__bf16*)carve((size_t)LBLK * CC * WID * WID * 2);
    __bf16* locwP  = (__bf16*)carve((size_t)LBLK * WID * WID * 2);
    __bf16* p1P    = (__bf16*)carve((size_t)WID * 2 * WID * 2);
    __bf16* p2P    = (__bf16*)carve((size_t)2 * WID * OUTD * 2);

    // 1. patch ranges
    k_ranges<<<1, 256, 0, stream>>>(batch, start, inv);

    // 2. pack weights into B-fragment order (bf16)
    k_pack_b<<<dim3((WID * WID + 255) / 256, LBLK * CC), 256, 0, stream>>>(
        Wc, WcP, WID, WID);
    k_pack_b<<<dim3((WID * WID + 255) / 256, LBLK), 256, 0, stream>>>(
        loc_w, locwP, WID, WID);
    k_pack_b<<<dim3((WID * 2 * WID + 255) / 256, 1), 256, 0, stream>>>(
        p1_w, p1P, WID, 2 * WID);
    k_pack_b<<<dim3((2 * WID * OUTD + 255) / 256, 1), 256, 0, stream>>>(
        p2_w, p2P, 2 * WID, OUTD);

    // 3. lift
    k_lift<<<(NPTS * WID + 255) / 256, 256, 0, stream>>>(x, lift_w, lift_b, v_a);

    // 4. kernel MLPs: phi row-major (N,C); psi transposed (C,N)
    for (int i = 0; i < LBLK; ++i) {
        k_mlp<<<NPTS / 128, 128, 0, stream>>>(
            x, phi_w1 + (size_t)i * DXD * HH, phi_b1 + (size_t)i * HH,
            phi_w2 + (size_t)i * HH * CC, phi_b2 + (size_t)i * CC,
            phiB + (size_t)i * NPTS * CC, CC, 1);
        k_mlp<<<NPTS / 128, 128, 0, stream>>>(
            x, psi_w1 + (size_t)i * DXD * HH, psi_b1 + (size_t)i * HH,
            psi_w2 + (size_t)i * HH * CC, psi_b2 + (size_t)i * CC,
            psiT + (size_t)i * CC * NPTS, 1, NPTS);
    }

    // 5. conv blocks
    __bf16* vc = v_a;
    __bf16* vn = v_b;
    for (int i = 0; i < LBLK; ++i) {
        k_seg_outer<<<PP, 256, 0, stream>>>(psiT + (size_t)i * CC * NPTS, vc,
                                            start, inv, sbuf);
        k_chan_gemm<<<dim3(PP / 16, CC), 256, 0, stream>>>(
            sbuf, WcP + (size_t)i * CC * WID * WID, tbuf);
        k_block_update<<<dim3(PP, WID / 16), 256, 0, stream>>>(
            phiB + (size_t)i * NPTS * CC, tbuf, vc,
            locwP + (size_t)i * WID * WID, loc_b + (size_t)i * WID, start, vn,
            (i < LBLK - 1) ? 1 : 0);
        __bf16* tmp = vc; vc = vn; vn = tmp;
    }

    // 6. pooling + projection head
    k_pool<<<PP, WID, 0, stream>>>(vc, start, inv, pooled);
    k_proj1<<<dim3(PP / 16, (2 * WID) / 16), 32, 0, stream>>>(pooled, p1P, p1_b,
                                                              hbuf);
    k_proj2<<<dim3(PP / 16, OUTD / 16), 32, 0, stream>>>(hbuf, p2P, p2_b, out);
}
